// AdditiveAttention_54082228191915
// MI455X (gfx1250) — matrix operand order repaired and checked by reading, not compiled, not run on hardware
//
#include <hip/hip_runtime.h>
#include <hip/hip_bf16.h>

typedef __attribute__((ext_vector_type(16))) __bf16 v16bf;
typedef __attribute__((ext_vector_type(8)))  float  v8f;

#define BATCH 64
#define SEQ   2048
#define ENCD  1024
#define DECD  1024
#define ATTN  512
#define NEGV  (-1e10f)

#define AROW  264   // 256 K + 8 bf16 pad per LDS row (bank-conflict avoidance)

__device__ __forceinline__ unsigned short f2bf(float f) {
    unsigned int u = __float_as_uint(f);
    u += 0x7fffu + ((u >> 16) & 1u);      // round-to-nearest-even
    return (unsigned short)(u >> 16);
}
__device__ __forceinline__ unsigned int pk2bf(float lo, float hi) {
    return (unsigned int)f2bf(lo) | ((unsigned int)f2bf(hi) << 16);
}

// gfx1250 hardware tanh (TRANS op). v_nop satisfies the TRANS-op
// result-use hazard (ISA 7.4.3) since the compiler can't see into asm.
__device__ __forceinline__ float tanh_hw(float x) {
    float r;
    asm("v_tanh_f32 %0, %1\n\tv_nop" : "=v"(r) : "v"(x));
    return r;
}

// ---------------------------------------------------------------------------
// Kernel 0: straight f32 -> bf16 conversion of W_enc (row-major [a][k] kept).
// Row-major bf16 serves as this code's B layout (same contiguous K order as A): lane reads 16 contiguous K
// from row a = nt*16 + lane%16 (+16 K for lanes 16..31).
// ---------------------------------------------------------------------------
__global__ __launch_bounds__(256) void k_convert_w(const float* __restrict__ W,
                                                   unsigned short* __restrict__ Wz) {
    int i4 = blockIdx.x * 256 + threadIdx.x;        // 0 .. 131071 float4s
    float4 f = ((const float4*)W)[i4];
    uint2 p;
    p.x = pk2bf(f.x, f.y);
    p.y = pk2bf(f.z, f.w);
    ((uint2*)Wz)[i4] = p;
}

// ---------------------------------------------------------------------------
// Kernel 1: proj_dec[b][a] = sum_k dec[b][k] * W_dec[a][k]   (64x512, tiny)
// ---------------------------------------------------------------------------
__global__ __launch_bounds__(256) void k_proj_dec(const float* __restrict__ dec,
                                                  const float* __restrict__ Wd,
                                                  float* __restrict__ pd) {
    int idx = blockIdx.x * 256 + threadIdx.x;       // 0 .. 32767
    int b = idx >> 9;
    int a = idx & 511;
    const float4* dv = (const float4*)(dec + (size_t)b * DECD);
    const float4* wv = (const float4*)(Wd  + (size_t)a * DECD);
    float s = 0.f;
    for (int i = 0; i < DECD / 4; ++i) {
        float4 x = dv[i], y = wv[i];
        s += x.x * y.x + x.y * y.y + x.z * y.z + x.w * y.w;
    }
    pd[idx] = s;
}

// ---------------------------------------------------------------------------
// Kernel 2: fused  scores[b][s] = v . tanh(enc[b,s,:]@W_enc^T + proj_dec[b,:])
// Block = 256 thr (8 waves) covers 64 S-rows of one batch.
//   wave w owns N-tiles nt = 4w..4w+3; 4 K-stages of 256.
//   A staged in LDS row-major bf16 (padded); per ktl: 4 A tiles shared by
//   all 4 N-tiles -> 16 WMMAs per ktl, LDS reads cut 4x.
// ---------------------------------------------------------------------------
__global__ __launch_bounds__(256) void k_scores(const float* __restrict__ enc,
                                                const unsigned short* __restrict__ Wz,
                                                const float* __restrict__ pd,
                                                const float* __restrict__ vvec,
                                                float* __restrict__ scores) {
    __shared__ __align__(16) unsigned short lds_a[64 * AROW];  // 33 KB
    __shared__ float red[8 * 64];

    const int tid  = threadIdx.x;
    const int wv   = tid >> 5;
    const int lane = tid & 31;
    const int b    = blockIdx.y;
    const int s0   = blockIdx.x * 64;

    v8f acc[4][4];
    for (int i = 0; i < 4; ++i)
        for (int mt = 0; mt < 4; ++mt)
            for (int e = 0; e < 8; ++e) acc[i][mt][e] = 0.f;

    const int m  = lane & 15;         // matrix row / column within tile
    const int hi = lane >> 4;         // half-wave select

    const float* srcb = enc + ((size_t)b * SEQ + s0) * ENCD;
    // B base: row a = wv*64 + m, upper half-wave offset +16 K
    const unsigned short* bbase = Wz + ((size_t)(wv * 64 + m)) * ENCD + hi * 16;

    union AV { uint4 u[2]; v16bf v; };

    for (int ks = 0; ks < 4; ++ks) {
        // ---- cooperative vectorized A fill: 64 rows x 256 K (bf16) ----
        for (int i = 0; i < 16; ++i) {
            int idx4 = i * 256 + tid;                 // float4 index, 0..4095
            int row  = idx4 >> 6;                     // 64 float4 per row
            int c4   = idx4 & 63;
            float4 f = *(const float4*)(srcb + (size_t)row * ENCD + ks * 256 + c4 * 4);
            uint2 p;
            p.x = pk2bf(f.x, f.y);
            p.y = pk2bf(f.z, f.w);
            *(uint2*)&lds_a[row * AROW + c4 * 4] = p;
        }
        __syncthreads();

#pragma unroll 1
        for (int ktl = 0; ktl < 8; ++ktl) {
            // A tiles: lane reads K chunk [K0..K0+15],
            // K0 = ktl*32 + hi*16, row = mt*16 + m
            AV av[4];
            for (int mt = 0; mt < 4; ++mt) {
                int off = (mt * 16 + m) * AROW + ktl * 32 + hi * 16;
                av[mt].u[0] = *(const uint4*)&lds_a[off];
                av[mt].u[1] = *(const uint4*)&lds_a[off + 8];
            }
            // B tiles: 32B contiguous per lane from L2
            v16bf bv[4];
            for (int ntl = 0; ntl < 4; ++ntl)
                bv[ntl] = *(const v16bf*)(bbase + (size_t)ntl * 16 * ENCD
                                          + ks * 256 + ktl * 32);
            for (int ntl = 0; ntl < 4; ++ntl)
                for (int mt = 0; mt < 4; ++mt)
                    acc[ntl][mt] = __builtin_amdgcn_wmma_f32_16x16x32_bf16(
                        false, av[mt].v, false, bv[ntl], (short)0,
                        acc[ntl][mt], false, false);
        }
        __syncthreads();
    }

    // ---- epilogue: + proj_dec, tanh (hw), * v, reduce over N ----
    float sc[4][8];
    for (int mt = 0; mt < 4; ++mt)
        for (int r = 0; r < 8; ++r) sc[mt][r] = 0.f;

    for (int ntl = 0; ntl < 4; ++ntl) {
        int n = (wv * 4 + ntl) * 16 + m;
        float pdv = pd[(size_t)b * ATTN + n];
        float vv  = vvec[n];
        for (int mt = 0; mt < 4; ++mt)
            for (int r = 0; r < 8; ++r)
                sc[mt][r] += tanh_hw(acc[ntl][mt][r] + pdv) * vv;
    }
    for (int mt = 0; mt < 4; ++mt)
        for (int r = 0; r < 8; ++r) {
            float x = sc[mt][r];
            x += __shfl_xor(x, 1);
            x += __shfl_xor(x, 2);
            x += __shfl_xor(x, 4);
            x += __shfl_xor(x, 8);
            sc[mt][r] = x;
        }
    if (m == 0) {
        for (int mt = 0; mt < 4; ++mt)
            for (int r = 0; r < 8; ++r)
                red[wv * 64 + mt * 16 + hi * 8 + r] = sc[mt][r];
    }
    __syncthreads();
    if (tid < 64) {
        float s = 0.f;
        for (int w = 0; w < 8; ++w) s += red[w * 64 + tid];
        scores[(size_t)b * SEQ + s0 + tid] = s;
    }
}

// ---------------------------------------------------------------------------
// Kernel 3: masked softmax over S per batch row; writes attn to d_out region.
// ---------------------------------------------------------------------------
__global__ __launch_bounds__(256) void k_softmax(const float* __restrict__ scores,
                                                 const int* __restrict__ mask,
                                                 float* __restrict__ attn) {
    __shared__ float red[256];
    int b = blockIdx.x, tid = threadIdx.x;
    float val[8];
    float mx = -3.4e38f;
    for (int i = 0; i < 8; ++i) {
        int s = i * 256 + tid;
        float x = (mask[(size_t)b * SEQ + s] == 0) ? NEGV
                                                   : scores[(size_t)b * SEQ + s];
        val[i] = x;
        mx = fmaxf(mx, x);
    }
    red[tid] = mx;
    __syncthreads();
    for (int off = 128; off > 0; off >>= 1) {
        if (tid < off) red[tid] = fmaxf(red[tid], red[tid + off]);
        __syncthreads();
    }
    mx = red[0];
    __syncthreads();
    float sum = 0.f;
    for (int i = 0; i < 8; ++i) {
        val[i] = __expf(val[i] - mx);
        sum += val[i];
    }
    red[tid] = sum;
    __syncthreads();
    for (int off = 128; off > 0; off >>= 1) {
        if (tid < off) red[tid] += red[tid + off];
        __syncthreads();
    }
    float inv = 1.f / red[0];
    for (int i = 0; i < 8; ++i)
        attn[(size_t)b * SEQ + i * 256 + tid] = val[i] * inv;
}

// ---------------------------------------------------------------------------
// Kernel 4/5: context = attn . enc  (deterministic two-pass, no fp atomics)
// ---------------------------------------------------------------------------
__global__ __launch_bounds__(256) void k_ctx_partial(const float* __restrict__ enc,
                                                     const float* __restrict__ attn,
                                                     float* __restrict__ part) {
    int c = blockIdx.x, b = blockIdx.y, tid = threadIdx.x;
    int e = tid * 4;
    float4 acc = {0.f, 0.f, 0.f, 0.f};
    const float* base = enc + ((size_t)b * SEQ + (size_t)c * 256) * ENCD;
    const float* ab   = attn + (size_t)b * SEQ + (size_t)c * 256;
    for (int s = 0; s < 256; ++s) {
        float a = ab[s];
        float4 ev = *(const float4*)(base + (size_t)s * ENCD + e);
        acc.x += a * ev.x; acc.y += a * ev.y;
        acc.z += a * ev.z; acc.w += a * ev.w;
    }
    *(float4*)(part + ((size_t)(c * BATCH + b)) * ENCD + e) = acc;
}

__global__ __launch_bounds__(256) void k_ctx_reduce(const float* __restrict__ part,
                                                    float* __restrict__ ctx) {
    int idx = blockIdx.x * 256 + threadIdx.x;   // 0 .. 65535 == b*1024 + e
    float s = 0.f;
    for (int c = 0; c < 8; ++c) s += part[(size_t)c * BATCH * ENCD + idx];
    ctx[idx] = s;
}

// ---------------------------------------------------------------------------
extern "C" void kernel_launch(void* const* d_in, const int* in_sizes, int n_in,
                              void* d_out, int out_size, void* d_ws, size_t ws_size,
                              hipStream_t stream) {
    const float* dec   = (const float*)d_in[0];
    const float* enc   = (const float*)d_in[1];
    const int*   mask  = (const int*)d_in[2];
    const float* W_enc = (const float*)d_in[3];
    const float* W_dec = (const float*)d_in[4];
    const float* vvec  = (const float*)d_in[5];

    float* out  = (float*)d_out;
    float* ctx  = out;                    // (64, 1024)
    float* attn = out + BATCH * ENCD;     // (64, 2048)

    // workspace layout (bytes): pd | scores | Wz(bf16) | partials  ~3.7 MB
    char* ws = (char*)d_ws;
    float*          pd     = (float*)(ws + 0);          // 64*512*4   = 131072
    float*          scores = (float*)(ws + 131072);     // 64*2048*4  = 524288
    unsigned short* Wz     = (unsigned short*)(ws + 655360);   // 512*1024*2 = 1 MB
    float*          part   = (float*)(ws + 1703936);    // 8*64*1024*4 = 2 MB

    k_convert_w<<<512, 256, 0, stream>>>(W_enc, Wz);
    k_proj_dec<<<128, 256, 0, stream>>>(dec, W_dec, pd);

    dim3 g2(SEQ / 64, BATCH);             // 32 x 64 = 2048 blocks
    k_scores<<<g2, 256, 0, stream>>>(enc, Wz, pd, vvec, scores);

    k_softmax<<<BATCH, 256, 0, stream>>>(scores, mask, attn);

    dim3 g4(8, BATCH);
    k_ctx_partial<<<g4, 256, 0, stream>>>(enc, attn, part);
    k_ctx_reduce<<<256, 256, 0, stream>>>(part, ctx);
}
